// FEconvModuleFE_14121852470127
// MI455X (gfx1250) — matrix operand order repaired and checked by reading, not compiled, not run on hardware
//
#include <hip/hip_runtime.h>
#include <hip/hip_bf16.h>

#define NZ 160
#define NY 160
#define NX 160
#define TILE_Y 8                                   // y-rows per block (1 wave each)
#define TILE_Z 4                                   // z-planes per wave (4 WMMAs)
#define STRIPS_X (NX / 16)                         // 10
#define NBLOCKS (STRIPS_X * (NY / TILE_Y) * (NZ / TILE_Z))   // 10*20*40 = 8000

// LDS halo tile: [6 z][10 y][18 x]
#define HZ (TILE_Z + 2)
#define HY (TILE_Y + 2)
#define HX 18
#define HP (HY * HX)                               // 180 (one z-plane)
#define HN (HZ * HP)                               // 1080

#define NTYPES 256
#define KPAD 32

typedef _Float16 v16h __attribute__((ext_vector_type(16)));
typedef _Float16 v8h  __attribute__((ext_vector_type(8)));
typedef _Float16 v2h  __attribute__((ext_vector_type(2)));
typedef float    v8f  __attribute__((ext_vector_type(8)));
typedef float    v2f  __attribute__((ext_vector_type(2)));

#if __has_builtin(__builtin_amdgcn_s_wait_asynccnt)
#define WAIT_ASYNC() __builtin_amdgcn_s_wait_asynccnt(0)
#else
#define WAIT_ASYNC() asm volatile("s_wait_asynccnt 0x0" ::: "memory")
#endif

// Filter prep: f32 [256][27] -> f16 [256][32] padded, so B fragments are two b128 loads.
__global__ __launch_bounds__(256) void prep_filters_kernel(
    const float* __restrict__ F, _Float16* __restrict__ Fh)
{
    const int i = blockIdx.x * 256 + threadIdx.x;   // 0 .. 8191
    const int t = i >> 5;
    const int k = i & 31;
    Fh[i] = (k < 27) ? (_Float16)F[t * 27 + k] : (_Float16)0.0f;
}

// Per wave: 4 z-planes of one 16-node x-strip. For each plane, D = A x B, diag(D).
__global__ __launch_bounds__(256) void fe_conv_wmma_kernel(
    const float* __restrict__ U,
    const int* __restrict__ types,
    const _Float16* __restrict__ Fh,
    float* __restrict__ out)
{
    __shared__ __align__(16) float    sRaw[HN];
    __shared__ __align__(16) _Float16 sH[HN];

    const int tid  = threadIdx.x;
    const int lane = tid & 31;
    const int w    = tid >> 5;         // wave id in block = y-row

    const int b  = blockIdx.x;
    const int x0 = (b % STRIPS_X) << 4;
    const int y0 = ((b / STRIPS_X) % (NY / TILE_Y)) * TILE_Y;
    const int z0 = (b / (STRIPS_X * (NY / TILE_Y))) * TILE_Z;

    // ---- Halo staging: threads 0..179 own one (y,x) column, loop 6 z-planes.
    // In-bounds -> CDNA5 async global->LDS copy; out-of-bounds -> zero fill.
    if (tid < HP) {
        const int dyi = tid / HX;
        const int dxi = tid % HX;
        const int yy = y0 - 1 + dyi;
        const int xx = x0 - 1 + dxi;
        const bool inplane = ((unsigned)yy < NY) & ((unsigned)xx < NX);
        #pragma unroll
        for (int dzi = 0; dzi < HZ; ++dzi) {
            const int zz = z0 - 1 + dzi;
            const int i = dzi * HP + tid;
            const unsigned ldsoff = (unsigned)(unsigned long long)(const void*)&sRaw[i];
            if (inplane && (unsigned)zz < NZ) {
                const float* gp = U + ((size_t)zz * NY + yy) * NX + xx;
                asm volatile("global_load_async_to_lds_b32 %0, %1, off"
                             :: "v"(ldsoff), "v"(gp) : "memory");
            } else {
                sRaw[i] = 0.0f;
            }
        }
    }
    WAIT_ASYNC();
    __syncthreads();

    // ---- In-LDS f32 -> f16, two elements at a time -------------------------
    {
        const v2f* pr = (const v2f*)sRaw;
        v2h*       ph = (v2h*)sH;
        for (int i = tid; i < HN / 2; i += 256) {
            const v2f f = pr[i];
            v2h h;
            h.x = (_Float16)f.x;
            h.y = (_Float16)f.y;
            ph[i] = h;
        }
    }
    __syncthreads();

    const int y    = y0 + w;
    const int half = lane >> 4;        // K-half owned by this lane
    const int mn   = lane & 15;        // A row / B column owned by this lane
    const int laneBase = w * HX + mn;  // (y-row, x) part of the LDS index

    #pragma unroll 1
    for (int dz4 = 0; dz4 < TILE_Z; ++dz4) {
        const int z = z0 + dz4;
        const int baseA = dz4 * HP + laneBase;

        // ---- A fragment: element e <-> K = e + 8*((e>>3) + half) ----------
        // Both candidate taps per element are compile-time; select with half.
        v16h a{};
        #pragma unroll
        for (int e = 0; e < 16; ++e) {
            const int k0  = e + ((e >= 8) ? 8 : 0);                 // half=0 tap (<27 always)
            const int k1  = k0 + 8;                                 // half=1 tap
            const int o0  = (k0 / 9) * HP + ((k0 / 3) % 3) * HX + (k0 % 3);
            const int o1  = (k1 < 27)
                          ? (k1 / 9) * HP + ((k1 / 3) % 3) * HX + (k1 % 3)
                          : o0;                                     // dummy (zeroed below)
            _Float16 v = sH[baseA + (half ? o1 : o0)];
            if (k1 >= 27)
                v = half ? (_Float16)0.0f : v;
            a[e] = v;
        }

        // ---- B fragment: two contiguous 16B loads from padded f16 table ---
        const int t = types[((size_t)z * NY + y) * NX + x0 + mn];
        const v8h* bp = (const v8h*)(Fh + ((t << 5) + (half << 3)));
        const v8h blo = bp[0];
        const v8h bhi = bp[2];
        const v16h b2 = __builtin_shufflevector(blo, bhi,
            0, 1, 2, 3, 4, 5, 6, 7, 8, 9, 10, 11, 12, 13, 14, 15);

        // ---- WMMA (f16 inputs, f32 accumulate) ----------------------------
        v8f c{};
        c = __builtin_amdgcn_wmma_f32_16x16x32_f16(
            false, a, false, b2, (short)0, c, false, false);

        // ---- diag(D): m<8 -> lane m, c[m]; m>=8 -> lane m+16, c[m-8] ------
        const bool lo = (lane < 8);
        const bool hi = (lane >= 24);
        if (lo || hi) {
            const int m = lo ? lane : (lane - 16);
            const int r = lo ? lane : (lane - 24);
            float v = 0.0f;
            #pragma unroll
            for (int i = 0; i < 8; ++i)
                if (i == r) v = c[i];
            out[((size_t)z * NY + y) * NX + x0 + m] = v;
        }
    }
}

extern "C" void kernel_launch(void* const* d_in, const int* in_sizes, int n_in,
                              void* d_out, int out_size, void* d_ws, size_t ws_size,
                              hipStream_t stream) {
    const float* U     = (const float*)d_in[0];
    const int*   types = (const int*)d_in[1];
    const float* F     = (const float*)d_in[2];
    float*       out   = (float*)d_out;
    _Float16*    Fh    = (_Float16*)d_ws;        // 16 KB scratch
    (void)in_sizes; (void)n_in; (void)out_size; (void)ws_size;

    prep_filters_kernel<<<(NTYPES * KPAD) / 256, 256, 0, stream>>>(F, Fh);
    fe_conv_wmma_kernel<<<NBLOCKS, 256, 0, stream>>>(U, types, Fh, out);
}
